// DeepSeekMoE_89945205113200
// MI455X (gfx1250) — matrix operand order repaired and checked by reading, not compiled, not run on hardware
//
#include <hip/hip_runtime.h>
#include <hip/hip_bf16.h>

// ---------------- problem constants (match reference) ----------------
#define T_TOK    4096
#define DMODEL   1024
#define DHIDDEN  1024
#define ETOT     64
#define KROUTED  6
#define KSHARED  2

// ---------------- GEMM tiling ----------------
#define BM 128          // block M tile
#define BN 64           // block N tile
#define BK 32           // K strip (one WMMA K)
#define LDA 40          // LDS row stride in bf16 elems (80B, multiple of 16B)
#define BUF_ELEMS ((BM + BN) * LDA)       // bf16 elems per pipeline buffer
#define BUF_BYTES (BUF_ELEMS * 2)

typedef __attribute__((ext_vector_type(16))) __bf16 v16bf;
typedef __attribute__((ext_vector_type(8)))  __bf16 v8bf;
typedef __attribute__((ext_vector_type(8)))  float  v8f;

__device__ __forceinline__ __bf16 f2bf(float f) {
  union { float f; unsigned u; } v; v.f = f;
  unsigned r = (v.u + 0x7FFFu + ((v.u >> 16) & 1u)) >> 16;
  unsigned short s = (unsigned short)r;
  __bf16 o; __builtin_memcpy(&o, &s, 2); return o;
}

__device__ __forceinline__ int imin(int a, int b) { return a < b ? a : b; }

// ---- async 16B copy global -> LDS (ASYNCcnt tracked). lds_addr = flat addr low 32 bits.
__device__ __forceinline__ void async_b128(unsigned lds_addr, unsigned long long gaddr) {
  asm volatile("global_load_async_to_lds_b128 %0, %1, off"
               :: "v"(lds_addr), "v"(gaddr) : "memory");
}
__device__ __forceinline__ void wait_async3() {
  asm volatile("s_wait_asynccnt 0x3" ::: "memory");
}
__device__ __forceinline__ void wait_async0() {
  asm volatile("s_wait_asynccnt 0x0" ::: "memory");
}

// A fragment: 16x32 bf16. lane: M = lane&15, hi = lane>>4.
// elem i<8 -> K = hi*8 + i ; elem i>=8 -> K = 16 + hi*8 + (i-8)
__device__ __forceinline__ v16bf load_a_frag(const __bf16* row, int hi) {
  v8bf lo = *(const v8bf*)(row + hi * 8);
  v8bf up = *(const v8bf*)(row + 16 + hi * 8);
  v16bf a;
#pragma unroll
  for (int i = 0; i < 8; ++i) { a[i] = lo[i]; a[8 + i] = up[i]; }
  return a;
}
// B fragment: 32x16 bf16. lane: N = lane&15, hi = lane>>4; elem i<8 -> K = hi*8 + i ; elem i>=8 -> K = 16 + hi*8 + (i-8).
__device__ __forceinline__ v16bf load_b_frag(const __bf16* colptr, int hi) {
  v8bf lo = *(const v8bf*)(colptr + hi * 8);
  v8bf up = *(const v8bf*)(colptr + 16 + hi * 8);
  v16bf b;
#pragma unroll
  for (int i = 0; i < 8; ++i) { b[i] = lo[i]; b[8 + i] = up[i]; }
  return b;
}

#define WMMA_BF16(A, B, C) \
  __builtin_amdgcn_wmma_f32_16x16x32_bf16(false, (A), false, (B), (short)0, (C), false, false)

// ---------------------------------------------------------------------------
// Shared GEMM core: C[BM x BN] += A * Bt^T over K, double-buffered async LDS.
//  A (!GATHER): points at row m0 of a [*, K] bf16 row-major matrix
//  A (GATHER):  base of full activation matrix; gidx[r] gives source row
//  Bt: points at row n0 of a [N, K] bf16 row-major (pre-transposed weight)
// ---------------------------------------------------------------------------
template <bool GATHER>
__device__ __forceinline__ void gemm_tiles(__bf16* smem,
                                           const __bf16* __restrict__ A,
                                           const int* __restrict__ gidx,
                                           int valid,
                                           const __bf16* __restrict__ Bt,
                                           int K, int tid, v8f acc[2][2]) {
  // per-thread 16B chunk assignment: A tile = 512 chunks (2/thread), B tile = 256 (1/thread)
  const int c2 = tid + 256;
  const int r1 = tid >> 2, kp1 = (tid & 3) * 8;
  const int r2 = c2 >> 2,  kp2 = (c2 & 3) * 8;
  const int bn = tid >> 2, kpb = (tid & 3) * 8;

  size_t rowA1, rowA2;
  if (GATHER) {
    rowA1 = (size_t)((r1 < valid) ? gidx[r1] : gidx[0]);
    rowA2 = (size_t)((r2 < valid) ? gidx[r2] : gidx[0]);
  } else {
    rowA1 = (size_t)((r1 < valid) ? r1 : 0);
    rowA2 = (size_t)((r2 < valid) ? r2 : 0);
  }
  const unsigned long long gA1 = (unsigned long long)(uintptr_t)(A + rowA1 * K + kp1);
  const unsigned long long gA2 = (unsigned long long)(uintptr_t)(A + rowA2 * K + kp2);
  const unsigned long long gB  = (unsigned long long)(uintptr_t)(Bt + (size_t)bn * K + kpb);

  const unsigned sbase = (unsigned)(uintptr_t)smem;   // LDS aperture: low 32 bits = LDS addr
  const unsigned lA1 = sbase + (unsigned)(r1 * LDA + kp1) * 2u;
  const unsigned lA2 = sbase + (unsigned)(r2 * LDA + kp2) * 2u;
  const unsigned lB  = sbase + (unsigned)((BM + bn) * LDA + kpb) * 2u;

  const int lane = tid & 31, hi = lane >> 4, nc = lane & 15;
  const int wv = tid >> 5, wm = wv & 3, wn = wv >> 2;

  // prologue: stage 0 into buffer 0
  async_b128(lA1, gA1);
  async_b128(lA2, gA2);
  async_b128(lB, gB);

  const int KS = K / BK;
  for (int s = 0; s < KS; ++s) {
    const int cur = s & 1;
    if (s + 1 < KS) {
      const unsigned boff = (unsigned)((cur ^ 1) * BUF_BYTES);
      const unsigned long long koff = (unsigned long long)(s + 1) * (BK * 2);
      async_b128(lA1 + boff, gA1 + koff);
      async_b128(lA2 + boff, gA2 + koff);
      async_b128(lB + boff, gB + koff);
      wait_async3();        // stage s landed (per-wave in-order, 3 issued per stage)
    } else {
      wait_async0();
    }
    __syncthreads();

    __bf16* bufp = smem + cur * BUF_ELEMS;
    const __bf16 (*lX)[LDA] = (const __bf16(*)[LDA])bufp;
    const __bf16 (*lW)[LDA] = (const __bf16(*)[LDA])(bufp + BM * LDA);
    v16bf a0 = load_a_frag(&lX[wm * 32 + 0  + nc][0], hi);
    v16bf a1 = load_a_frag(&lX[wm * 32 + 16 + nc][0], hi);
    v16bf b0 = load_b_frag(&lW[wn * 32 + 0  + nc][0], hi);
    v16bf b1 = load_b_frag(&lW[wn * 32 + 16 + nc][0], hi);
    acc[0][0] = WMMA_BF16(a0, b0, acc[0][0]);
    acc[0][1] = WMMA_BF16(a0, b1, acc[0][1]);
    acc[1][0] = WMMA_BF16(a1, b0, acc[1][0]);
    acc[1][1] = WMMA_BF16(a1, b1, acc[1][1]);
    __syncthreads();        // all reads of buf[cur] done before it is refilled
  }
}

// ---------------- kernel: zero per-expert counters ----------------
__global__ void moe_zero_counts(int* __restrict__ counts) {
  counts[threadIdx.x] = 0;
}

// ---------------- kernel: gate logits + top-6 + softmax + route ----------------
__global__ __launch_bounds__(64)
void moe_gate_topk(const float* __restrict__ u, const float* __restrict__ gw,
                   int* __restrict__ counts, int* __restrict__ tok_idx,
                   float* __restrict__ tok_w) {
  const int t = blockIdx.x;
  const int e = threadIdx.x;
  __shared__ float lg[ETOT];
  const float* ur = u + (size_t)t * DMODEL;
  float acc = 0.f;
#pragma unroll 4
  for (int d = 0; d < DMODEL; ++d)
    acc += ur[d] * gw[(size_t)d * ETOT + e];
  lg[e] = acc;
  __syncthreads();
  if (e == 0) {
    unsigned long long used = 0ull;
    float vals[KROUTED]; int ids[KROUTED];
    for (int j = 0; j < KROUTED; ++j) {
      float best = -3.4e38f; int bi = 0;
      for (int q = 0; q < ETOT; ++q)
        if (!((used >> q) & 1ull) && lg[q] > best) { best = lg[q]; bi = q; }
      used |= 1ull << bi; vals[j] = best; ids[j] = bi;
    }
    float mx = vals[0], s = 0.f, ex[KROUTED];
    for (int j = 0; j < KROUTED; ++j) { ex[j] = __expf(vals[j] - mx); s += ex[j]; }
    float inv = 1.f / s;
    for (int j = 0; j < KROUTED; ++j) {
      int ee = ids[j];
      int pos = atomicAdd(&counts[ee], 1);
      tok_idx[(size_t)ee * T_TOK + pos] = t;
      tok_w[(size_t)ee * T_TOK + pos] = ex[j] * inv;
    }
  }
}

// ---------------- kernel: exclusive scan of counts ----------------
__global__ void moe_scan_offsets(const int* __restrict__ counts, int* __restrict__ offsets) {
  if (threadIdx.x == 0) {
    int s = 0;
    for (int e = 0; e < ETOT; ++e) { offsets[e] = s; s += counts[e]; }
  }
}

// ---------------- kernel: fp32 -> bf16 (activations) ----------------
__global__ __launch_bounds__(256)
void moe_convert_act(const float* __restrict__ src, __bf16* __restrict__ dst) {
  const size_t i = ((size_t)blockIdx.x * 256 + threadIdx.x) * 4;
  float4 v = *(const float4*)(src + i);
  dst[i + 0] = f2bf(v.x); dst[i + 1] = f2bf(v.y);
  dst[i + 2] = f2bf(v.z); dst[i + 3] = f2bf(v.w);
}

// ---------------- kernel: fp32 [z][R][C] -> bf16 transposed [z][C][R] ----------------
#define TT 32
__global__ __launch_bounds__(256)
void moe_convert_transpose(const float* __restrict__ src, __bf16* __restrict__ dst,
                           int R, int C) {
  const int z = blockIdx.z;
  src += (size_t)z * R * C;
  dst += (size_t)z * R * C;
  __shared__ __bf16 tile[TT][TT + 2];
  const int bx = blockIdx.x * TT;   // C offset
  const int by = blockIdx.y * TT;   // R offset
  const int tx = threadIdx.x & 31, ty8 = threadIdx.x >> 5;
  for (int rr = ty8; rr < TT; rr += 8)
    tile[rr][tx] = f2bf(src[(size_t)(by + rr) * C + bx + tx]);
  __syncthreads();
  for (int rr = ty8; rr < TT; rr += 8)
    dst[(size_t)(bx + rr) * R + by + tx] = tile[tx][rr];
}

// ---------------- kernel: shared FFN1  hs = relu(ub @ Ws1 + bs1) ----------------
__global__ __launch_bounds__(256)
void moe_shared_ffn1(const __bf16* __restrict__ ub, const __bf16* __restrict__ Wt,
                     const float* __restrict__ bs1, __bf16* __restrict__ hs) {
  const int e = blockIdx.z;
  const int m0 = blockIdx.y * BM;
  const int n0 = blockIdx.x * BN;
  __shared__ __bf16 smem[2 * BUF_ELEMS];
  const int tid = threadIdx.x;
  v8f acc[2][2] = {};
  gemm_tiles<false>(smem, ub + (size_t)m0 * DMODEL, nullptr, BM,
                    Wt + (size_t)e * DMODEL * DHIDDEN + (size_t)n0 * DMODEL,
                    DMODEL, tid, acc);
  const int lane = tid & 31, hi = lane >> 4, nc = lane & 15;
  const int wv = tid >> 5, wm = wv & 3, wn = wv >> 2;
#pragma unroll
  for (int ms = 0; ms < 2; ++ms)
#pragma unroll
    for (int r = 0; r < 8; ++r) {
      int m = m0 + wm * 32 + ms * 16 + hi * 8 + r;
#pragma unroll
      for (int ns = 0; ns < 2; ++ns) {
        int col = n0 + wn * 32 + ns * 16 + nc;
        float hv = acc[ms][ns][r] + bs1[(size_t)e * DHIDDEN + col];
        hv = hv > 0.f ? hv : 0.f;
        hs[(size_t)e * T_TOK * DHIDDEN + (size_t)m * DHIDDEN + col] = f2bf(hv);
      }
    }
}

// ---------------- kernel: shared FFN2 + writeout ----------------
__global__ __launch_bounds__(256)
void moe_shared_ffn2_out(const __bf16* __restrict__ hs, const __bf16* __restrict__ Wt,
                         const float* __restrict__ bs2, const float* __restrict__ u,
                         float* __restrict__ out) {
  const int m0 = blockIdx.y * BM;
  const int n0 = blockIdx.x * BN;
  __shared__ __bf16 smem[2 * BUF_ELEMS];
  const int tid = threadIdx.x;
  v8f acc[2][2] = {};
  for (int e = 0; e < KSHARED; ++e)
    gemm_tiles<false>(smem, hs + (size_t)e * T_TOK * DHIDDEN + (size_t)m0 * DHIDDEN,
                      nullptr, BM,
                      Wt + (size_t)e * DHIDDEN * DMODEL + (size_t)n0 * DHIDDEN,
                      DHIDDEN, tid, acc);
  const int lane = tid & 31, hi = lane >> 4, nc = lane & 15;
  const int wv = tid >> 5, wm = wv & 3, wn = wv >> 2;
#pragma unroll
  for (int ms = 0; ms < 2; ++ms)
#pragma unroll
    for (int r = 0; r < 8; ++r) {
      int m = m0 + wm * 32 + ms * 16 + hi * 8 + r;
#pragma unroll
      for (int ns = 0; ns < 2; ++ns) {
        int col = n0 + wn * 32 + ns * 16 + nc;
        float v = (acc[ms][ns][r] + bs2[col] + bs2[DMODEL + col]) * (1.f / KSHARED);
        out[(size_t)m * DMODEL + col] = u[(size_t)m * DMODEL + col] + v;
      }
    }
}

// ---------------- kernel: routed FFN1 (token-gathered) ----------------
__global__ __launch_bounds__(256)
void moe_routed_ffn1(const __bf16* __restrict__ ub, const __bf16* __restrict__ Wt,
                     const float* __restrict__ br1, const int* __restrict__ counts,
                     const int* __restrict__ offsets, const int* __restrict__ tok_idx,
                     __bf16* __restrict__ hr) {
  const int e = blockIdx.z;
  const int cnt = counts[e];
  const int m0 = blockIdx.y * BM;
  if (m0 >= cnt) return;
  const int n0 = blockIdx.x * BN;
  const int off = offsets[e];
  const int valid = imin(cnt - m0, BM);
  __shared__ __bf16 smem[2 * BUF_ELEMS];
  const int tid = threadIdx.x;
  v8f acc[2][2] = {};
  gemm_tiles<true>(smem, ub, tok_idx + (size_t)e * T_TOK + m0, valid,
                   Wt + (size_t)e * DMODEL * DHIDDEN + (size_t)n0 * DMODEL,
                   DMODEL, tid, acc);
  const int lane = tid & 31, hi = lane >> 4, nc = lane & 15;
  const int wv = tid >> 5, wm = wv & 3, wn = wv >> 2;
#pragma unroll
  for (int ms = 0; ms < 2; ++ms)
#pragma unroll
    for (int r = 0; r < 8; ++r) {
      int m = m0 + wm * 32 + ms * 16 + hi * 8 + r;
      if (m >= cnt) continue;
#pragma unroll
      for (int ns = 0; ns < 2; ++ns) {
        int col = n0 + wn * 32 + ns * 16 + nc;
        float hv = acc[ms][ns][r] + br1[(size_t)e * DHIDDEN + col];
        hv = hv > 0.f ? hv : 0.f;
        hr[(size_t)(off + m) * DHIDDEN + col] = f2bf(hv);
      }
    }
}

// ---------------- kernel: routed FFN2 + weighted scatter ----------------
__global__ __launch_bounds__(256)
void moe_routed_ffn2(const __bf16* __restrict__ hr, const __bf16* __restrict__ Wt,
                     const float* __restrict__ br2, const int* __restrict__ counts,
                     const int* __restrict__ offsets, const int* __restrict__ tok_idx,
                     const float* __restrict__ tok_w, float* __restrict__ out) {
  const int e = blockIdx.z;
  const int cnt = counts[e];
  const int m0 = blockIdx.y * BM;
  if (m0 >= cnt) return;
  const int n0 = blockIdx.x * BN;
  const int off = offsets[e];
  const int valid = imin(cnt - m0, BM);
  __shared__ __bf16 smem[2 * BUF_ELEMS];
  const int tid = threadIdx.x;
  v8f acc[2][2] = {};
  gemm_tiles<false>(smem, hr + (size_t)(off + m0) * DHIDDEN, nullptr, valid,
                    Wt + (size_t)e * DHIDDEN * DMODEL + (size_t)n0 * DHIDDEN,
                    DHIDDEN, tid, acc);
  const int lane = tid & 31, hi = lane >> 4, nc = lane & 15;
  const int wv = tid >> 5, wm = wv & 3, wn = wv >> 2;
#pragma unroll
  for (int ms = 0; ms < 2; ++ms)
#pragma unroll
    for (int r = 0; r < 8; ++r) {
      int m = m0 + wm * 32 + ms * 16 + hi * 8 + r;
      if (m >= cnt) continue;
      int tok = tok_idx[(size_t)e * T_TOK + m];
      float wgt = tok_w[(size_t)e * T_TOK + m];
#pragma unroll
      for (int ns = 0; ns < 2; ++ns) {
        int col = n0 + wn * 32 + ns * 16 + nc;
        float yv = acc[ms][ns][r] + br2[(size_t)e * DMODEL + col];
        atomicAdd(&out[(size_t)tok * DMODEL + col], wgt * yv);
      }
    }
}

// ---------------- host launcher ----------------
extern "C" void kernel_launch(void* const* d_in, const int* in_sizes, int n_in,
                              void* d_out, int out_size, void* d_ws, size_t ws_size,
                              hipStream_t stream) {
  (void)in_sizes; (void)n_in; (void)out_size; (void)ws_size;
  const float* u   = (const float*)d_in[0];
  const float* gw  = (const float*)d_in[1];
  const float* Ws1 = (const float*)d_in[2];
  const float* bs1 = (const float*)d_in[3];
  const float* Ws2 = (const float*)d_in[4];
  const float* bs2 = (const float*)d_in[5];
  const float* Wr1 = (const float*)d_in[6];
  const float* br1 = (const float*)d_in[7];
  const float* Wr2 = (const float*)d_in[8];
  const float* br2 = (const float*)d_in[9];
  float* out = (float*)d_out;

  // workspace carve-up (~338 MB)
  char* ws = (char*)d_ws;
  int*    counts  = (int*)ws;    ws += 256;
  int*    offsets = (int*)ws;    ws += 256;
  int*    tok_idx = (int*)ws;    ws += sizeof(int)    * (size_t)ETOT * T_TOK;
  float*  tok_w   = (float*)ws;  ws += sizeof(float)  * (size_t)ETOT * T_TOK;
  __bf16* ub      = (__bf16*)ws; ws += sizeof(__bf16) * (size_t)T_TOK * DMODEL;
  __bf16* hs      = (__bf16*)ws; ws += sizeof(__bf16) * (size_t)KSHARED * T_TOK * DHIDDEN;
  __bf16* hr      = (__bf16*)ws; ws += sizeof(__bf16) * (size_t)KROUTED * T_TOK * DHIDDEN;
  __bf16* Wts1    = (__bf16*)ws; ws += sizeof(__bf16) * (size_t)KSHARED * DMODEL * DHIDDEN;
  __bf16* Wts2    = (__bf16*)ws; ws += sizeof(__bf16) * (size_t)KSHARED * DHIDDEN * DMODEL;
  __bf16* Wtr1    = (__bf16*)ws; ws += sizeof(__bf16) * (size_t)ETOT * DMODEL * DHIDDEN;
  __bf16* Wtr2    = (__bf16*)ws; // ETOT * DHIDDEN * DMODEL bf16

  // routing chain
  moe_zero_counts<<<1, ETOT, 0, stream>>>(counts);
  moe_gate_topk<<<T_TOK, ETOT, 0, stream>>>(u, gw, counts, tok_idx, tok_w);
  moe_scan_offsets<<<1, 32, 0, stream>>>(counts, offsets);

  // precision / layout pre-pass: bf16 activations, bf16 transposed weights
  moe_convert_act<<<(T_TOK * DMODEL) / (256 * 4), 256, 0, stream>>>(u, ub);
  moe_convert_transpose<<<dim3(DHIDDEN / TT, DMODEL / TT, KSHARED), 256, 0, stream>>>(Ws1, Wts1, DMODEL, DHIDDEN);
  moe_convert_transpose<<<dim3(DMODEL / TT, DHIDDEN / TT, KSHARED), 256, 0, stream>>>(Ws2, Wts2, DHIDDEN, DMODEL);
  moe_convert_transpose<<<dim3(DHIDDEN / TT, DMODEL / TT, ETOT), 256, 0, stream>>>(Wr1, Wtr1, DMODEL, DHIDDEN);
  moe_convert_transpose<<<dim3(DMODEL / TT, DHIDDEN / TT, ETOT), 256, 0, stream>>>(Wr2, Wtr2, DHIDDEN, DMODEL);

  // shared experts
  moe_shared_ffn1<<<dim3(DHIDDEN / BN, T_TOK / BM, KSHARED), 256, 0, stream>>>(ub, Wts1, bs1, hs);
  moe_shared_ffn2_out<<<dim3(DMODEL / BN, T_TOK / BM), 256, 0, stream>>>(hs, Wts2, bs2, u, out);

  // routed experts (token-gathered, early-exit grid)
  moe_routed_ffn1<<<dim3(DHIDDEN / BN, T_TOK / BM, ETOT), 256, 0, stream>>>(
      ub, Wtr1, br1, counts, offsets, tok_idx, hr);
  moe_routed_ffn2<<<dim3(DMODEL / BN, T_TOK / BM, ETOT), 256, 0, stream>>>(
      hr, Wtr2, br2, counts, offsets, tok_idx, tok_w, out);
}
